// Decoder_1262720385183
// MI455X (gfx1250) — matrix operand order repaired and checked by reading, not compiled, not run on hardware
//
#include <hip/hip_runtime.h>

#define DEVI __device__ __forceinline__

constexpr int Bc  = 2;
constexpr int NRc = 16384;
constexpr int NPc = 65536;
constexpr int Ec  = 262144;
constexpr int Fc  = 128;

typedef __attribute__((ext_vector_type(16))) __bf16 v16bf;
typedef __attribute__((ext_vector_type(8)))  float  v8f;
typedef __attribute__((ext_vector_type(4)))  unsigned u32x4;
struct Pack32 { u32x4 lo, hi; };   // 32 bytes, trivially copyable

// ---------- scalar helpers ----------
DEVI unsigned short f2bf(float x) {           // round-to-nearest-even fp32->bf16
  unsigned u = __builtin_bit_cast(unsigned, x);
  u += 0x7FFFu + ((u >> 16) & 1u);
  return (unsigned short)(u >> 16);
}
DEVI float swishf(float x) { return x / (1.f + __expf(-x)); }
DEVI void atomAdd(float* p, float v) {
  __hip_atomic_fetch_add(p, v, __ATOMIC_RELAXED, __HIP_MEMORY_SCOPE_AGENT);
}

// ---------- WMMA fragment loaders (CDNA5 wave32 layouts, CDNA5 ISA) ----------
// A 16x32 bf16: lane m=L&15; j<8 -> K = kb + j, j>=8 -> K = kb + 16 + (j-8),
// kb = ks*32 + (L>=16 ? 8 : 0). Two contiguous 16B chunks per lane.
DEVI v16bf ldsA(const unsigned short* base, int pitch, int ks, int lane) {
  int off = (lane & 15) * pitch + ks * 32 + ((lane & 16) ? 8 : 0);
  Pack32 p;
  p.lo = *reinterpret_cast<const u32x4*>(base + off);
  p.hi = *reinterpret_cast<const u32x4*>(base + off + 16);
  return __builtin_bit_cast(v16bf, p);
}
// B 32x16 bf16 fragments pre-swizzled in global: frag = nt*(K/32)+ks,
// element j of lane L stored contiguously -> one 32B coalesced load per lane.
DEVI v16bf gblB(const unsigned short* frags, int fragIdx, int lane) {
  const u32x4* p = reinterpret_cast<const u32x4*>(frags + ((fragIdx * 32 + lane) << 4));
  Pack32 q; q.lo = p[0]; q.hi = p[1];
  return __builtin_bit_cast(v16bf, q);
}
DEVI v8f wmma_bf16(v16bf a, v16bf b, v8f c) {
  return __builtin_amdgcn_wmma_f32_16x16x32_bf16(false, a, false, b, (short)0, c, false, false);
}
template<int KS>
DEVI v8f gemm_col(const unsigned short* A, int apitch, const unsigned short* Wf, int nt, int lane) {
  v8f acc = {0.f,0.f,0.f,0.f,0.f,0.f,0.f,0.f};
#pragma unroll
  for (int ks = 0; ks < KS; ++ks)
    acc = wmma_bf16(ldsA(A, apitch, ks, lane), gblB(Wf, nt * KS + ks, lane), acc);
  return acc;
}

// ---------- prep kernels ----------
__global__ void zero_kernel(float* p, int n) {
  int i = blockIdx.x * blockDim.x + threadIdx.x;
  int s = gridDim.x * blockDim.x;
  for (; i < n; i += s) p[i] = 0.f;
}

// swizzle W[K][128] fp32 -> bf16 WMMA-B fragment order
__global__ void wfrag_kernel(const float* __restrict__ W, unsigned short* __restrict__ out, int K) {
  int tid = blockIdx.x * blockDim.x + threadIdx.x;
  int total = K * 128;
  if (tid >= total) return;
  int j = tid & 15, lane = (tid >> 4) & 31, frag = tid >> 9;
  int KS = K >> 5, ks = frag % KS, nt = frag / KS;
  int k = ks * 32 + ((lane & 16) ? 8 : 0) + (j & 7) + ((j & 8) ? 16 : 0);
  int n = nt * 16 + (lane & 15);
  out[tid] = f2bf(W[k * 128 + n]);
}

// cond-norm vectors: h = swish(tau*C1 + c1)[16]; sc/sh = h@C2 + c2 (split)
__global__ void cond_kernel(const float* __restrict__ tau,
                            const float* __restrict__ C1, const float* __restrict__ c1,
                            const float* __restrict__ C2, const float* __restrict__ c2,
                            float* __restrict__ scale, float* __restrict__ shift) {
  int b = blockIdx.x, f = threadIdx.x;      // f in [0,256)
  float t = tau[b];
  float acc = c2[f];
  for (int i = 0; i < 16; ++i) {
    float hv = swishf(t * C1[i] + c1[i]);
    acc += hv * C2[i * 256 + f];
  }
  if (f < 128) scale[b * 128 + f] = 1.f + acc;
  else         shift[b * 128 + (f - 128)] = acc;
}

// ---------- edge kernel: embed MLP + cond-LN + edge-update MLP + cond-LN + atomic segment-sum ----------
constexpr int APE = 392;  // concat [16][384] pitch (x2B = 784B: 16B-aligned, conflict-free)
constexpr int HPI = 136;  // hidden [16][128] bf16 pitch
constexpr int FPI = 132;  // fp32 [16][128] pitch
constexpr int EW_BYTES = 16*APE*2 + 16*HPI*2 + 16*FPI*4 + 16*FPI*4 + 256; // 34048/wave

__global__ __launch_bounds__(128) void edge_kernel(
    const float* __restrict__ rnode, const float* __restrict__ pnode,
    const float* __restrict__ efeat,
    const int* __restrict__ senders, const int* __restrict__ recvs,
    const float* __restrict__ We1, const float* __restrict__ be1,
    const unsigned short* __restrict__ We2f, const float* __restrict__ be2,
    const unsigned short* __restrict__ Wu1f, const float* __restrict__ bu1,
    const unsigned short* __restrict__ Wu2f, const float* __restrict__ bu2,
    const float* __restrict__ scE, const float* __restrict__ shE,
    const float* __restrict__ scU, const float* __restrict__ shU,
    float* __restrict__ agg, float* __restrict__ cnt) {
  extern __shared__ char smem[];
  const int lane = threadIdx.x & 31, wv = threadIdx.x >> 5;
  const int tilesPerBatch = Ec / 64;
  const int b  = blockIdx.x / tilesPerBatch;
  const int e0 = (blockIdx.x % tilesPerBatch) * 64 + wv * 16;

  char* w0 = smem + wv * EW_BYTES;
  unsigned short* aB = (unsigned short*)w0;                       // [16][392] bf16 concat
  unsigned short* hB = (unsigned short*)(w0 + 16*APE*2);          // [16][136] bf16 hidden
  float* fb = (float*)(w0 + 16*APE*2 + 16*HPI*2);                 // [16][132] f32 gemm out
  float* ef = (float*)(w0 + 16*APE*2 + 16*HPI*2 + 16*FPI*4);      // [16][132] f32 e value
  int*   sidx = (int*)(w0 + 16*APE*2 + 16*HPI*2 + 2*16*FPI*4);
  int*   ridx = sidx + 16;
  float* mu = (float*)(ridx + 16);
  float* rs = mu + 16;

  if (lane < 16) {
    sidx[lane] = senders[b * Ec + e0 + lane];
    ridx[lane] = recvs  [b * Ec + e0 + lane];
  }
  // layer1 (K=4) in VALU: h1 = swish(ef@We1 + be1)
  for (int idx = lane; idx < 16 * Fc; idx += 32) {
    int m = idx >> 7, f = idx & 127;
    const float* x = efeat + (long)(b * Ec + e0 + m) * 4;
    float v = be1[f] + x[0]*We1[f] + x[1]*We1[128+f] + x[2]*We1[256+f] + x[3]*We1[384+f];
    hB[m * HPI + f] = f2bf(swishf(v));
  }
  __syncthreads();
  // GEMM2: fb = h1 @ We2 + be2  (WMMA, K=128)
  {
    int n15 = lane & 15, mb = (lane & 16) ? 8 : 0;
    for (int nt = 0; nt < 8; ++nt) {
      v8f acc = gemm_col<4>(hB, HPI, We2f, nt, lane);
      int n = nt * 16 + n15;
#pragma unroll
      for (int r = 0; r < 8; ++r) fb[(mb + r) * FPI + n] = acc[r] + be2[n];
    }
  }
  __syncthreads();
  if (lane < 16) {                       // LN stats per row
    float s = 0.f, s2 = 0.f;
    for (int f = 0; f < Fc; ++f) { float v = fb[lane * FPI + f]; s += v; s2 += v * v; }
    float m_ = s * (1.f / Fc);
    mu[lane] = m_; rs[lane] = rsqrtf(s2 * (1.f / Fc) - m_ * m_ + 1e-6f);
  }
  __syncthreads();
  {
    const float* sc = scE + b * Fc; const float* sh = shE + b * Fc;
    for (int idx = lane; idx < 16 * Fc; idx += 32) {   // e = LN*scale+shift
      int m = idx >> 7, f = idx & 127;
      float v = (fb[m * FPI + f] - mu[m]) * rs[m] * sc[f] + sh[f];
      ef[m * FPI + f] = v;
      aB[m * APE + f] = f2bf(v);
    }
  }
  // gather sender/receiver features (bf16 into concat cols 128..383)
  for (int idx = lane; idx < 16 * Fc; idx += 32) {
    int m = idx >> 7, f = idx & 127;
    aB[m * APE + 128 + f] = f2bf(rnode[((long)b * NRc + sidx[m]) * Fc + f]);
    aB[m * APE + 256 + f] = f2bf(pnode[((long)b * NPc + ridx[m]) * Fc + f]);
  }
  __syncthreads();
  // GEMM3: hB = swish(concat @ Wu1 + bu1)  (WMMA, K=384)
  {
    int n15 = lane & 15, mb = (lane & 16) ? 8 : 0;
    for (int nt = 0; nt < 8; ++nt) {
      v8f acc = gemm_col<12>(aB, APE, Wu1f, nt, lane);
      int n = nt * 16 + n15;
#pragma unroll
      for (int r = 0; r < 8; ++r) hB[(mb + r) * HPI + n] = f2bf(swishf(acc[r] + bu1[n]));
    }
  }
  __syncthreads();
  // GEMM4: fb = hB @ Wu2 + bu2
  {
    int n15 = lane & 15, mb = (lane & 16) ? 8 : 0;
    for (int nt = 0; nt < 8; ++nt) {
      v8f acc = gemm_col<4>(hB, HPI, Wu2f, nt, lane);
      int n = nt * 16 + n15;
#pragma unroll
      for (int r = 0; r < 8; ++r) fb[(mb + r) * FPI + n] = acc[r] + bu2[n];
    }
  }
  __syncthreads();
  if (lane < 16) {
    float s = 0.f, s2 = 0.f;
    for (int f = 0; f < Fc; ++f) { float v = fb[lane * FPI + f]; s += v; s2 += v * v; }
    float m_ = s * (1.f / Fc);
    mu[lane] = m_; rs[lane] = rsqrtf(s2 * (1.f / Fc) - m_ * m_ + 1e-6f);
  }
  __syncthreads();
  {
    const float* sc = scU + b * Fc; const float* sh = shU + b * Fc;
    for (int idx = lane; idx < 16 * Fc; idx += 32) {   // e += LN(update); scatter to agg
      int m = idx >> 7, f = idx & 127;
      float v = ef[m * FPI + f] + (fb[m * FPI + f] - mu[m]) * rs[m] * sc[f] + sh[f];
      atomAdd(&agg[((long)b * NPc + ridx[m]) * Fc + f], v);
    }
    if (lane < 16) atomAdd(&cnt[(long)b * NPc + ridx[lane]], 1.0f);
  }
}

// ---------- pnode kernel: segment-mean + pnode MLP + cond-LN + residual + output MLP ----------
constexpr int APP = 264;  // concat [16][256] pitch
constexpr int PW_BYTES = 16*APP*2 + 16*HPI*2 + 16*FPI*4 + 16*FPI*4 + 256; // 29952/wave

__global__ __launch_bounds__(128) void pnode_kernel(
    const float* __restrict__ pnode, const float* __restrict__ agg, const float* __restrict__ cnt,
    const unsigned short* __restrict__ Wp1f, const float* __restrict__ bp1,
    const unsigned short* __restrict__ Wp2f, const float* __restrict__ bp2,
    const float* __restrict__ scP, const float* __restrict__ shP,
    const unsigned short* __restrict__ Wo1f, const float* __restrict__ bo1,
    const float* __restrict__ Wo2, const float* __restrict__ bo2,
    float* __restrict__ out) {
  extern __shared__ char smem[];
  const int lane = threadIdx.x & 31, wv = threadIdx.x >> 5;
  const int tilesPerBatch = NPc / 64;
  const int b  = blockIdx.x / tilesPerBatch;
  const int n0 = (blockIdx.x % tilesPerBatch) * 64 + wv * 16;

  char* w0 = smem + wv * PW_BYTES;
  unsigned short* aB = (unsigned short*)w0;                       // [16][264] bf16 concat
  unsigned short* hB = (unsigned short*)(w0 + 16*APP*2);          // [16][136]
  float* fb = (float*)(w0 + 16*APP*2 + 16*HPI*2);                 // [16][132]
  float* ef = (float*)(w0 + 16*APP*2 + 16*HPI*2 + 16*FPI*4);      // [16][132] p residual
  float* mu = (float*)(w0 + 16*APP*2 + 16*HPI*2 + 2*16*FPI*4);
  float* rs = mu + 16;

  for (int idx = lane; idx < 16 * Fc; idx += 32) {
    int m = idx >> 7, f = idx & 127;
    long ni = (long)b * NPc + n0 + m;
    float pv = pnode[ni * Fc + f];
    ef[m * FPI + f] = pv;
    aB[m * APP + f] = f2bf(pv);
    float c = fmaxf(cnt[ni], 1.f);
    aB[m * APP + 128 + f] = f2bf(agg[ni * Fc + f] / c);
  }
  __syncthreads();
  {  // GEMM Wp1 (K=256) -> swish -> hB
    int n15 = lane & 15, mb = (lane & 16) ? 8 : 0;
    for (int nt = 0; nt < 8; ++nt) {
      v8f acc = gemm_col<8>(aB, APP, Wp1f, nt, lane);
      int n = nt * 16 + n15;
#pragma unroll
      for (int r = 0; r < 8; ++r) hB[(mb + r) * HPI + n] = f2bf(swishf(acc[r] + bp1[n]));
    }
  }
  __syncthreads();
  {  // GEMM Wp2 -> fb
    int n15 = lane & 15, mb = (lane & 16) ? 8 : 0;
    for (int nt = 0; nt < 8; ++nt) {
      v8f acc = gemm_col<4>(hB, HPI, Wp2f, nt, lane);
      int n = nt * 16 + n15;
#pragma unroll
      for (int r = 0; r < 8; ++r) fb[(mb + r) * FPI + n] = acc[r] + bp2[n];
    }
  }
  __syncthreads();
  if (lane < 16) {
    float s = 0.f, s2 = 0.f;
    for (int f = 0; f < Fc; ++f) { float v = fb[lane * FPI + f]; s += v; s2 += v * v; }
    float m_ = s * (1.f / Fc);
    mu[lane] = m_; rs[lane] = rsqrtf(s2 * (1.f / Fc) - m_ * m_ + 1e-6f);
  }
  __syncthreads();
  {
    const float* sc = scP + b * Fc; const float* sh = shP + b * Fc;
    for (int idx = lane; idx < 16 * Fc; idx += 32) {  // p_new = p + LN*scale+shift
      int m = idx >> 7, f = idx & 127;
      float v = ef[m * FPI + f] + (fb[m * FPI + f] - mu[m]) * rs[m] * sc[f] + sh[f];
      aB[m * APP + f] = f2bf(v);
    }
  }
  __syncthreads();
  {  // GEMM Wo1 -> swish fp32 into fb
    int n15 = lane & 15, mb = (lane & 16) ? 8 : 0;
    for (int nt = 0; nt < 8; ++nt) {
      v8f acc = gemm_col<4>(aB, APP, Wo1f, nt, lane);
      int n = nt * 16 + n15;
#pragma unroll
      for (int r = 0; r < 8; ++r) fb[(mb + r) * FPI + n] = swishf(acc[r] + bo1[n]);
    }
  }
  __syncthreads();
  for (int idx = lane; idx < 64; idx += 32) {  // final 128 -> 4 in VALU
    int m = idx >> 2, o = idx & 3;
    float acc = bo2[o];
    for (int f = 0; f < Fc; ++f) acc += fb[m * FPI + f] * Wo2[f * 4 + o];
    out[((long)b * NPc + n0 + m) * 4 + o] = acc;
  }
}

// ---------- host ----------
extern "C" void kernel_launch(void* const* d_in, const int* in_sizes, int n_in,
                              void* d_out, int out_size, void* d_ws, size_t ws_size,
                              hipStream_t stream) {
  const float* rnode = (const float*)d_in[0];
  const float* pnode = (const float*)d_in[1];
  const float* efeat = (const float*)d_in[2];
  const float* tau   = (const float*)d_in[3];
  const float* We1 = (const float*)d_in[4];  const float* We2 = (const float*)d_in[5];
  const float* be1 = (const float*)d_in[6];  const float* be2 = (const float*)d_in[7];
  const float* Ce1 = (const float*)d_in[8];  const float* Ce2 = (const float*)d_in[9];
  const float* ce1 = (const float*)d_in[10]; const float* ce2 = (const float*)d_in[11];
  const float* Wu1 = (const float*)d_in[12]; const float* Wu2 = (const float*)d_in[13];
  const float* bu1 = (const float*)d_in[14]; const float* bu2 = (const float*)d_in[15];
  const float* Cu1 = (const float*)d_in[16]; const float* Cu2 = (const float*)d_in[17];
  const float* cu1 = (const float*)d_in[18]; const float* cu2 = (const float*)d_in[19];
  const float* Wp1 = (const float*)d_in[20]; const float* Wp2 = (const float*)d_in[21];
  const float* bp1 = (const float*)d_in[22]; const float* bp2 = (const float*)d_in[23];
  const float* Cp1 = (const float*)d_in[24]; const float* Cp2 = (const float*)d_in[25];
  const float* cp1 = (const float*)d_in[26]; const float* cp2 = (const float*)d_in[27];
  // 28..35: rnode-update params (dead code in reference) -- skipped
  const float* Wo1 = (const float*)d_in[36]; const float* Wo2 = (const float*)d_in[37];
  const float* bo1 = (const float*)d_in[38]; const float* bo2 = (const float*)d_in[39];
  const int* senders = (const int*)d_in[40];
  const int* recvs   = (const int*)d_in[41];
  float* out = (float*)d_out;

  // workspace carve
  char* ws = (char*)d_ws;
  size_t off = 0;
  auto carve = [&](size_t bytes) { size_t r = off; off = (off + bytes + 255) & ~(size_t)255; return r; };
  size_t oWe2 = carve(128 * 128 * 2);
  size_t oWu1 = carve(384 * 128 * 2);
  size_t oWu2 = carve(128 * 128 * 2);
  size_t oWp1 = carve(256 * 128 * 2);
  size_t oWp2 = carve(128 * 128 * 2);
  size_t oWo1 = carve(128 * 128 * 2);
  size_t oScE = carve(Bc * 128 * 4); size_t oShE = carve(Bc * 128 * 4);
  size_t oScU = carve(Bc * 128 * 4); size_t oShU = carve(Bc * 128 * 4);
  size_t oScP = carve(Bc * 128 * 4); size_t oShP = carve(Bc * 128 * 4);
  size_t oAgg = carve((size_t)Bc * NPc * Fc * 4);
  size_t oCnt = carve((size_t)Bc * NPc * 4);
  (void)ws_size; (void)n_in; (void)in_sizes; (void)out_size;

  unsigned short* We2f = (unsigned short*)(ws + oWe2);
  unsigned short* Wu1f = (unsigned short*)(ws + oWu1);
  unsigned short* Wu2f = (unsigned short*)(ws + oWu2);
  unsigned short* Wp1f = (unsigned short*)(ws + oWp1);
  unsigned short* Wp2f = (unsigned short*)(ws + oWp2);
  unsigned short* Wo1f = (unsigned short*)(ws + oWo1);
  float* scE = (float*)(ws + oScE); float* shE = (float*)(ws + oShE);
  float* scU = (float*)(ws + oScU); float* shU = (float*)(ws + oShU);
  float* scP = (float*)(ws + oScP); float* shP = (float*)(ws + oShP);
  float* agg = (float*)(ws + oAgg);
  float* cnt = (float*)(ws + oCnt);

  // zero agg + cnt (adjacent)
  int nzero = Bc * NPc * Fc + Bc * NPc;
  zero_kernel<<<4096, 256, 0, stream>>>(agg, nzero);

  // weight swizzles
  wfrag_kernel<<<(128 * 128 + 255) / 256, 256, 0, stream>>>(We2, We2f, 128);
  wfrag_kernel<<<(384 * 128 + 255) / 256, 256, 0, stream>>>(Wu1, Wu1f, 384);
  wfrag_kernel<<<(128 * 128 + 255) / 256, 256, 0, stream>>>(Wu2, Wu2f, 128);
  wfrag_kernel<<<(256 * 128 + 255) / 256, 256, 0, stream>>>(Wp1, Wp1f, 256);
  wfrag_kernel<<<(128 * 128 + 255) / 256, 256, 0, stream>>>(Wp2, Wp2f, 128);
  wfrag_kernel<<<(128 * 128 + 255) / 256, 256, 0, stream>>>(Wo1, Wo1f, 128);

  // cond-norm vectors
  cond_kernel<<<Bc, 256, 0, stream>>>(tau, Ce1, ce1, Ce2, ce2, scE, shE);
  cond_kernel<<<Bc, 256, 0, stream>>>(tau, Cu1, cu1, Cu2, cu2, scU, shU);
  cond_kernel<<<Bc, 256, 0, stream>>>(tau, Cp1, cp1, Cp2, cp2, scP, shP);

  // allow big dynamic LDS (no-op if unnecessary; errors ignored)
  (void)hipFuncSetAttribute((const void*)edge_kernel,
      hipFuncAttributeMaxDynamicSharedMemorySize, EW_BYTES * 4);
  (void)hipFuncSetAttribute((const void*)pnode_kernel,
      hipFuncAttributeMaxDynamicSharedMemorySize, PW_BYTES * 4);

  // edge stage: 4 waves/block, 16 edges/wave
  edge_kernel<<<Bc * (Ec / 64), 128, EW_BYTES * 4, stream>>>(
      rnode, pnode, efeat, senders, recvs,
      We1, be1, We2f, be2, Wu1f, bu1, Wu2f, bu2,
      scE, shE, scU, shU, agg, cnt);

  // pnode stage
  pnode_kernel<<<Bc * (NPc / 64), 128, PW_BYTES * 4, stream>>>(
      pnode, agg, cnt, Wp1f, bp1, Wp2f, bp2, scP, shP,
      Wo1f, bo1, Wo2, bo2, out);
}
